// Map_2276332667427
// MI455X (gfx1250) — matrix operand order repaired and checked by reading, not compiled, not run on hardware
//
#include <hip/hip_runtime.h>
#include <hip/hip_bf16.h>

// ---------------------------------------------------------------------------
// Problem constants (from reference): B=256, F=64, H=W=64, D=256, HC=32
// ---------------------------------------------------------------------------
#define NB   256
#define NF   64
#define NH   64
#define NW   64
#define ND   256
#define NHC  32

typedef __attribute__((ext_vector_type(16))) __bf16 bf16x16;
typedef __attribute__((ext_vector_type(8)))  float  floatx8;

union BF16Frag {
    uint4   q[2];   // two 16-byte chunks = 16 bf16
    bf16x16 v;
};

__device__ __forceinline__ unsigned short f2bf(float f) {
    union { float f; unsigned u; } x; x.f = f;
    unsigned r = x.u + 0x7FFFu + ((x.u >> 16) & 1u);   // round-to-nearest-even
    return (unsigned short)(r >> 16);
}
__device__ __forceinline__ uint4 zero4() { uint4 z; z.x=z.y=z.z=z.w=0u; return z; }

__device__ __forceinline__ floatx8 wmma_bf16(const BF16Frag& a, const BF16Frag& b, floatx8 c) {
    return __builtin_amdgcn_wmma_f32_16x16x32_bf16(false, a.v, false, b.v, (short)0, c, false, false);
}

// ---------------------------------------------------------------------------
// 1) prep_state: state f32 [b][c][y][x]  ->  new_state copy (f32, d_out)
//                                         +  stateT bf16 [b][y][x][c]
// One block per (b, y). 256 threads. LDS 64x65 f32 transpose tile.
// ---------------------------------------------------------------------------
__global__ void prep_state(const float* __restrict__ state,
                           float* __restrict__ newstate,
                           unsigned short* __restrict__ stateT) {
    __shared__ float tile[64 * 65];
    const int b = blockIdx.x >> 6;
    const int y = blockIdx.x & 63;
    {
        const int x  = threadIdx.x & 63;
        const int c4 = threadIdx.x >> 6;      // 0..3
        #pragma unroll
        for (int cc = 0; cc < 16; ++cc) {
            int c = cc * 4 + c4;
            size_t gi = ((size_t)(b * NF + c) * NH + y) * NW + x;
            float v = state[gi];
            newstate[gi]   = v;               // fused 256MB copy
            tile[x * 65 + c] = v;
        }
    }
    __syncthreads();
    {
        const int c  = threadIdx.x & 63;
        const int x4 = threadIdx.x >> 6;
        unsigned short* orow = stateT + ((size_t)(b * NH + y) * NW) * NF;
        #pragma unroll
        for (int xx = 0; xx < 16; ++xx) {
            int xo = xx * 4 + x4;
            orow[xo * NF + c] = f2bf(tile[xo * 65 + c]);   // contiguous in c
        }
    }
}

// ---------------------------------------------------------------------------
// 2) prep_small: conv weights -> wB[oc][K] bf16 (K = (r*3+s)*IC + ic),
//                x -> bf16, write_w -> wwT[n][384] bf16, zero r_acc.
// ---------------------------------------------------------------------------
__global__ void prep_small(const float* __restrict__ c1w, const float* __restrict__ c2w,
                           const float* __restrict__ c3w, const float* __restrict__ xin,
                           const float* __restrict__ ww,
                           unsigned short* __restrict__ wB1, unsigned short* __restrict__ wB2,
                           unsigned short* __restrict__ wB3, unsigned short* __restrict__ xbf,
                           unsigned short* __restrict__ wwT, float* __restrict__ racc) {
    const int stride = gridDim.x * blockDim.x;
    const int tid = blockIdx.x * blockDim.x + threadIdx.x;
    for (int i = tid; i < NHC * 9 * NF; i += stride) {               // conv1: 32*576
        int oc = i / (9 * NF), k = i % (9 * NF);
        int t = k >> 6, ic = k & 63, r = t / 3, s = t % 3;
        wB1[i] = f2bf(c1w[((oc * NF + ic) * 3 + r) * 3 + s]);
    }
    for (int i = tid; i < NHC * 9 * NHC; i += stride) {              // conv2/3: 32*288
        int oc = i / (9 * NHC), k = i % (9 * NHC);
        int t = k >> 5, ic = k & 31, r = t / 3, s = t % 3;
        wB2[i] = f2bf(c2w[((oc * NHC + ic) * 3 + r) * 3 + s]);
        wB3[i] = f2bf(c3w[((oc * NHC + ic) * 3 + r) * 3 + s]);
    }
    for (int i = tid; i < NB * ND; i += stride) xbf[i] = f2bf(xin[i]);
    for (int i = tid; i < NF * (ND + 2 * NF); i += stride) {
        int n = i / (ND + 2 * NF), k = i % (ND + 2 * NF);
        wwT[i] = f2bf(ww[k * NF + n]);
    }
    for (int i = tid; i < NB * NF; i += stride) racc[i] = 0.f;
}

// ---------------------------------------------------------------------------
// 3) prep_linT: lin_w (131072 x 64) f32  ->  linT (64 x 131072) bf16
//    LDS-tiled 2D transpose; one block per 64 K-rows.
// ---------------------------------------------------------------------------
__global__ void prep_linT(const float* __restrict__ lin_w, unsigned short* __restrict__ linT) {
    __shared__ float tile[64 * 65];
    const int K0 = blockIdx.x * 64;
    {
        const int n = threadIdx.x & 63, kr = threadIdx.x >> 6;
        #pragma unroll
        for (int kk = 0; kk < 16; ++kk) {
            int kl = kk * 4 + kr;
            tile[kl * 65 + n] = lin_w[(size_t)(K0 + kl) * NF + n];   // coalesced
        }
    }
    __syncthreads();
    const int n2 = threadIdx.x >> 2;           // 0..63
    const int j0 = (threadIdx.x & 3) * 16;     // 16 K each
    unsigned int* dst = (unsigned int*)(linT + (size_t)n2 * (NHC * NH * NW) + K0 + j0);
    #pragma unroll
    for (int jj = 0; jj < 8; ++jj) {
        unsigned lo = f2bf(tile[(j0 + 2 * jj    ) * 65 + n2]);
        unsigned hi = f2bf(tile[(j0 + 2 * jj + 1) * 65 + n2]);
        dst[jj] = lo | (hi << 16);
    }
}

// ---------------------------------------------------------------------------
// 4) conv3x3_wmma<IC, MODE>: implicit-GEMM 3x3 SAME conv via WMMA bf16.
//    Input:  inT bf16 [b][y][x][IC]  (channel-last)
//    Weights: wB bf16 [oc][9*IC], K = (r*3+s)*IC + ic
//    MODE 0: out bf16 [b][y][x][32]  (LDS repack, coalesced b128 stores)
//    MODE 1: out bf16 [b][oc][y][x]  (direct packed b128 store per lane)
//    One block per (b, y): 8 waves = 4 pixel-tiles(16) x 2 oc-tiles(16).
//    Dual accumulators (by unrolled-step parity) break the WMMA C-chain so
//    consecutive v_wmma are independent and overlap with the next tap's loads.
// ---------------------------------------------------------------------------
template <int IC, int MODE>
__global__ void conv3x3_wmma(const unsigned short* __restrict__ inT,
                             const unsigned short* __restrict__ wB,
                             const float* __restrict__ bias,
                             unsigned short* __restrict__ out) {
    constexpr int KTOT = 9 * IC;
    const int b    = blockIdx.x >> 6;
    const int y    = blockIdx.x & 63;
    const int wave = threadIdx.x >> 5;
    const int lane = threadIdx.x & 31;
    const int half = lane >> 4;     // 0/1: which K-half this lane covers
    const int m    = lane & 15;     // row within tile / col within tile
    const int x0   = (wave & 3) * 16;          // pixel tile base
    const int oc   = (wave >> 2) * 16 + m;     // this lane's output channel

    const unsigned short* wrow = wB + oc * KTOT;
    floatx8 acc0 = {}, acc1 = {};

    #pragma unroll
    for (int t = 0; t < 9; ++t) {
        const int r = t / 3, s = t - r * 3;
        const int yy = y + r - 1;
        if (yy < 0 || yy >= NH) continue;                 // uniform over block
        const int xx = x0 + m + s - 1;
        const bool xok = (xx >= 0) & (xx < NW);           // per-lane zero-pad
        const unsigned short* arow =
            inT + ((size_t)(b * NH + yy) * NW + (xok ? xx : 0)) * IC;
        #pragma unroll
        for (int ch = 0; ch < IC / 32; ++ch) {
            const int k0 = t * IC + ch * 32;
            BF16Frag a, bw;
            if (xok) {   // A 16x32 bf16: lane<16 -> K {0..7,16..23}, lane>=16 -> {8..15,24..31}
                a.q[0] = *(const uint4*)(arow + ch * 32 + half * 8);
                a.q[1] = *(const uint4*)(arow + ch * 32 + 16 + half * 8);
            } else { a.q[0] = zero4(); a.q[1] = zero4(); }
            // B 32x16 bf16: lane<16 -> K {0..7,16..23}, lane>=16 -> {8..15,24..31} (col = lane&15)
            bw.q[0] = *(const uint4*)(wrow + k0 + half * 8);
            bw.q[1] = *(const uint4*)(wrow + k0 + 16 + half * 8);
            if (((t * (IC / 32) + ch) & 1) == 0) acc0 = wmma_bf16(a, bw, acc0);
            else                                 acc1 = wmma_bf16(a, bw, acc1);
        }
    }

    const float bv = bias[oc];
    float cv[8];
    #pragma unroll
    for (int v = 0; v < 8; ++v) {
        float t = acc0[v] + acc1[v] + bv;
        cv[v] = t > 0.f ? t : 0.f;
    }

    if (MODE == 0) {
        __shared__ unsigned short tile[NW * NHC];          // 4 KB
        #pragma unroll
        for (int v = 0; v < 8; ++v)                        // D row M = v + half*8
            tile[(x0 + half * 8 + v) * NHC + oc] = f2bf(cv[v]);
        __syncthreads();
        if (threadIdx.x < (NW * NHC) / 8) {                // 256 x 16B = whole row
            uint4* dst = (uint4*)(out + (size_t)(b * NH + y) * NW * NHC);
            dst[threadIdx.x] = ((const uint4*)tile)[threadIdx.x];
        }
    } else {
        uint4 p;   // 8 consecutive-x bf16 -> one b128 store
        p.x = (unsigned)f2bf(cv[0]) | ((unsigned)f2bf(cv[1]) << 16);
        p.y = (unsigned)f2bf(cv[2]) | ((unsigned)f2bf(cv[3]) << 16);
        p.z = (unsigned)f2bf(cv[4]) | ((unsigned)f2bf(cv[5]) << 16);
        p.w = (unsigned)f2bf(cv[6]) | ((unsigned)f2bf(cv[7]) << 16);
        *(uint4*)(out + ((size_t)(b * NHC + oc) * NH + y) * NW + x0 + half * 8) = p;
    }
}

// ---------------------------------------------------------------------------
// 5) linear_wmma: r_acc[b][n] += sum_K h3[b][K] * linT[n][K], split-K=64,
//    accumulate with global_atomic_add_f32. K = 131072 (matches reference
//    reshape order because h3 is stored [b][c][y][x]).
//    Manually unrolled 2x with independent accumulator chains: 8 b128 loads
//    per iteration in one clause, two independent v_wmma so the second
//    overlaps the first's wait shadow.
// ---------------------------------------------------------------------------
#define LIN_K   (NHC * NH * NW)    // 131072
#define LIN_SPL 64
__global__ void linear_wmma(const unsigned short* __restrict__ h3,
                            const unsigned short* __restrict__ linT,
                            float* __restrict__ racc) {
    const int job  = blockIdx.x * 8 + (threadIdx.x >> 5);   // 16 * 4 * 64 jobs
    const int lane = threadIdx.x & 31;
    const int half = lane >> 4;
    const int m    = lane & 15;
    const int mt = job & 15, nt = (job >> 4) & 3, sp = job >> 6;
    const unsigned short* arow = h3   + (size_t)(mt * 16 + m) * LIN_K;
    const unsigned short* brow = linT + (size_t)(nt * 16 + m) * LIN_K;
    floatx8 c0 = {}, c1 = {};
    const int kbeg = sp * (LIN_K / LIN_SPL), kend = kbeg + (LIN_K / LIN_SPL);
    for (int k0 = kbeg; k0 < kend; k0 += 64) {
        BF16Frag a0, b0, a1, b1;
        a0.q[0] = *(const uint4*)(arow + k0 + half * 8);
        a0.q[1] = *(const uint4*)(arow + k0 + 16 + half * 8);
        b0.q[0] = *(const uint4*)(brow + k0 + half * 8);
        b0.q[1] = *(const uint4*)(brow + k0 + 16 + half * 8);
        a1.q[0] = *(const uint4*)(arow + k0 + 32 + half * 8);
        a1.q[1] = *(const uint4*)(arow + k0 + 48 + half * 8);
        b1.q[0] = *(const uint4*)(brow + k0 + 32 + half * 8);
        b1.q[1] = *(const uint4*)(brow + k0 + 48 + half * 8);
        c0 = wmma_bf16(a0, b0, c0);
        c1 = wmma_bf16(a1, b1, c1);
    }
    const int n = nt * 16 + m;
    #pragma unroll
    for (int v = 0; v < 8; ++v) {
        int bb = mt * 16 + half * 8 + v;
        atomicAdd(racc + bb * NF + n, c0[v] + c1[v]);
    }
}

// ---------------------------------------------------------------------------
// 6) finalize_r: r = relu(acc + lin_b); write f32 to output head, bf16 to ws.
// ---------------------------------------------------------------------------
__global__ void finalize_r(const float* __restrict__ racc, const float* __restrict__ lin_b,
                           float* __restrict__ out_head, unsigned short* __restrict__ rbf) {
    const int i = blockIdx.x * 256 + threadIdx.x;          // < 16384
    const int b = i >> 6, c = i & 63;
    float v = racc[i] + lin_b[c];
    v = v > 0.f ? v : 0.f;
    out_head[b * 128 + c] = v;
    rbf[i] = f2bf(v);
}

// ---------------------------------------------------------------------------
// 7) write_wmma: w = [x | r | m] @ write_w + b  (K=384), WMMA;
//    m gathered from stateT (contiguous in c); scatter w into new_state
//    and write w into the output head. 64 wave-jobs = 8 blocks x 8 waves.
// ---------------------------------------------------------------------------
__global__ void write_wmma(const unsigned short* __restrict__ xbf,
                           const unsigned short* __restrict__ rbf,
                           const unsigned short* __restrict__ stateT,
                           const unsigned short* __restrict__ wwT,
                           const int* __restrict__ index,
                           const float* __restrict__ wbias,
                           float* __restrict__ out_head,
                           float* __restrict__ out_state) {
    const int job  = blockIdx.x * 8 + (threadIdx.x >> 5);
    const int lane = threadIdx.x & 31;
    const int half = lane >> 4;
    const int m    = lane & 15;
    const int mt = job & 15, nt = job >> 4;
    const int bb = mt * 16 + m;                 // A row: batch
    const int n  = nt * 16 + m;                 // B row: output dim
    const int i0 = index[2 * bb], i1 = index[2 * bb + 1];
    const unsigned short* mrow = stateT + ((size_t)(bb * NH + i0) * NW + i1) * NF;
    const unsigned short* xrow = xbf + bb * ND;
    const unsigned short* rrow = rbf + bb * NF;
    const unsigned short* brow = wwT + n * (ND + 2 * NF);
    floatx8 c = {};
    #pragma unroll
    for (int k0 = 0; k0 < ND + 2 * NF; k0 += 32) {
        const unsigned short* aseg; int ko;
        if (k0 < ND)           { aseg = xrow; ko = k0; }
        else if (k0 < ND + NF) { aseg = rrow; ko = k0 - ND; }
        else                   { aseg = mrow; ko = k0 - ND - NF; }
        BF16Frag a, b;
        a.q[0] = *(const uint4*)(aseg + ko + half * 8);
        a.q[1] = *(const uint4*)(aseg + ko + 16 + half * 8);
        b.q[0] = *(const uint4*)(brow + k0 + half * 8);
        b.q[1] = *(const uint4*)(brow + k0 + 16 + half * 8);
        c = wmma_bf16(a, b, c);
    }
    const float bv = wbias[n];
    #pragma unroll
    for (int v = 0; v < 8; ++v) {
        const int bo = mt * 16 + half * 8 + v;
        const float val = c[v] + bv;
        out_head[bo * 128 + NF + n] = val;
        const int j0 = index[2 * bo], j1 = index[2 * bo + 1];
        out_state[(size_t)(bo * NF + n) * (NH * NW) + j0 * NW + j1] = val;
    }
}

// ---------------------------------------------------------------------------
// Launch
// ---------------------------------------------------------------------------
extern "C" void kernel_launch(void* const* d_in, const int* in_sizes, int n_in,
                              void* d_out, int out_size, void* d_ws, size_t ws_size,
                              hipStream_t stream) {
    (void)in_sizes; (void)n_in; (void)out_size; (void)ws_size;
    const float* x     = (const float*)d_in[0];
    const float* state = (const float*)d_in[1];
    const int*   index = (const int*)d_in[2];
    const float* c1w = (const float*)d_in[3];
    const float* c1b = (const float*)d_in[4];
    const float* c2w = (const float*)d_in[5];
    const float* c2b = (const float*)d_in[6];
    const float* c3w = (const float*)d_in[7];
    const float* c3b = (const float*)d_in[8];
    const float* lin_w = (const float*)d_in[9];
    const float* lin_b = (const float*)d_in[10];
    const float* ww  = (const float*)d_in[11];
    const float* wb  = (const float*)d_in[12];

    float* out_head  = (float*)d_out;                 // (256,128) = [r | w]
    float* out_state = out_head + NB * 2 * NF;        // (256,64,64,64)

    // Workspace layout (bytes)
    char* ws = (char*)d_ws;
    size_t off = 0;
    unsigned short* stateT = (unsigned short*)(ws + off); off += (size_t)NB*NH*NW*NF*2;     // 134 MB
    unsigned short* hA     = (unsigned short*)(ws + off); off += (size_t)NB*NH*NW*NHC*2;    //  67 MB (h1, later h3)
    unsigned short* hB     = (unsigned short*)(ws + off); off += (size_t)NB*NH*NW*NHC*2;    //  67 MB (h2)
    unsigned short* linT   = (unsigned short*)(ws + off); off += (size_t)NF*LIN_K*2;        //  17 MB
    unsigned short* wB1    = (unsigned short*)(ws + off); off += 65536;
    unsigned short* wB2    = (unsigned short*)(ws + off); off += 65536;
    unsigned short* wB3    = (unsigned short*)(ws + off); off += 65536;
    unsigned short* xbf    = (unsigned short*)(ws + off); off += (size_t)NB*ND*2;
    unsigned short* wwT    = (unsigned short*)(ws + off); off += 65536;
    float*          racc   = (float*)(ws + off);          off += (size_t)NB*NF*4;
    unsigned short* rbf    = (unsigned short*)(ws + off); off += (size_t)NB*NF*2;

    const dim3 blk(256);

    // Prep: state copy+transpose, weight/x re-layouts, lin_w transpose
    prep_state<<<NB * NH, blk, 0, stream>>>(state, out_state, stateT);
    prep_small<<<128, blk, 0, stream>>>(c1w, c2w, c3w, x, ww, wB1, wB2, wB3, xbf, wwT, racc);
    prep_linT<<<LIN_K / 64, blk, 0, stream>>>(lin_w, linT);

    // Conv tower (all WMMA bf16)
    conv3x3_wmma<NF,  0><<<NB * NH, blk, 0, stream>>>(stateT, wB1, c1b, hA);  // h1 [b][y][x][32]
    conv3x3_wmma<NHC, 0><<<NB * NH, blk, 0, stream>>>(hA,     wB2, c2b, hB);  // h2 [b][y][x][32]
    conv3x3_wmma<NHC, 1><<<NB * NH, blk, 0, stream>>>(hB,     wB3, c3b, hA);  // h3 [b][c][y][x]

    // Linear (split-K WMMA + f32 atomics), then bias+relu
    linear_wmma<<<(16 * 4 * LIN_SPL) / 8, blk, 0, stream>>>(hA, linT, racc);
    finalize_r<<<(NB * NF) / 256, blk, 0, stream>>>(racc, lin_b, out_head, rbf);

    // Write GEMM + gather m + scatter into new_state
    write_wmma<<<8, blk, 0, stream>>>(xbf, rbf, stateT, wwT, index, wb, out_head, out_state);
}
